// GroupedQueryAttention_37976100831528
// MI455X (gfx1250) — compile-verified, matrix operand order checked by reading, not run on hardware
//
#include <hip/hip_runtime.h>
#include <stdint.h>

typedef __bf16 bf16_t;
typedef __attribute__((ext_vector_type(16))) __bf16 v16bf;
typedef __attribute__((ext_vector_type(8)))  float  v8f;

#define D_MODEL 2048
#define N_HEADS 32
#define N_KV    8
#define D_K     64
#define BATCH   2
#define SEQ     2048
#define MTOK    (BATCH * SEQ)          /* 4096 tokens */
#define KV_DIM  (N_KV * D_K)           /* 512 */

// ---------------------------------------------------------------------------
// fp32 -> bf16 round-to-nearest-even helpers
// ---------------------------------------------------------------------------
__device__ __forceinline__ uint16_t f2bfu(float x) {
  uint32_t u = __builtin_bit_cast(uint32_t, x);
  return (uint16_t)((u + 0x7FFFu + ((u >> 16) & 1u)) >> 16);
}
__device__ __forceinline__ uint32_t pack2bf(float x0, float x1) {
  return (uint32_t)f2bfu(x0) | ((uint32_t)f2bfu(x1) << 16);
}

// ---------------------------------------------------------------------------
// WMMA 16-bit operand fragment (outer=16 x K=32) from a row-major bf16 row.
// ISA layout (CDNA5 ISA): for lane-half h, this lane's 16 elements are
// exactly bytes [16h, 16h+16) and [32+16h, 32+16h+16) of its 64-byte row
// -> two b128 loads, zero VALU.
// ---------------------------------------------------------------------------
union FragU { v16bf f; uint4 u[2]; };

__device__ __forceinline__ v16bf load_frag_bf16(const bf16_t* rowptr, int half) {
  const uint4* p = (const uint4*)rowptr;
  FragU fu;
  fu.u[0] = p[half];
  fu.u[1] = p[2 + half];
  return fu.f;
}

#define WMMA_BF16(a, b, c) \
  __builtin_amdgcn_wmma_f32_16x16x32_bf16(false, (a), false, (b), (short)0, (c), false, false)

// ---------------------------------------------------------------------------
// fp32 -> bf16 bulk convert (4 elements / thread)
// ---------------------------------------------------------------------------
__global__ void __launch_bounds__(256)
cvt_f32_bf16_kernel(const float* __restrict__ src, uint32_t* __restrict__ dst, int n4) {
  int i = blockIdx.x * blockDim.x + threadIdx.x;
  if (i >= n4) return;
  float4 x = ((const float4*)src)[i];
  dst[2 * i]     = pack2bf(x.x, x.y);
  dst[2 * i + 1] = pack2bf(x.z, x.w);
}

// ---------------------------------------------------------------------------
// GEMM:  C[M,N] = A[M,K] * W[N,K]^T + bias[N]   (torch Linear convention)
// One wave computes a 32x64 strip: 8 accumulators, each B fragment reused
// twice. Output either fp32 (Cf) or bf16 (Cb); exactly one is non-null.
// ---------------------------------------------------------------------------
__global__ void __launch_bounds__(256)
gemm_bf16_kernel(const bf16_t* __restrict__ A, const bf16_t* __restrict__ W,
                 const float* __restrict__ bias,
                 float* __restrict__ Cf, uint16_t* __restrict__ Cb,
                 int M, int N, int K) {
  const int lane   = threadIdx.x & 31;
  const int wave   = threadIdx.x >> 5;
  const int tilesN = N >> 6;
  const int tile   = blockIdx.x * (blockDim.x >> 5) + wave;
  if (tile >= (M >> 5) * tilesN) return;
  const int m0   = (tile / tilesN) << 5;
  const int n0   = (tile % tilesN) << 6;
  const int nidx = lane & 15;
  const int half = (lane >> 4) & 1;

  v8f acc0[4] = {}, acc1[4] = {};
  const bf16_t* A0 = A + (size_t)(m0 + nidx) * K;
  const bf16_t* A1 = A + (size_t)(m0 + 16 + nidx) * K;
  const bf16_t* Wr[4];
#pragma unroll
  for (int t = 0; t < 4; ++t) Wr[t] = W + (size_t)(n0 + t * 16 + nidx) * K;

  for (int kb = 0; kb < K; kb += 32) {
    __builtin_prefetch(A0 + kb + 256, 0, 1);   // global_prefetch_b8, next tiles
    v16bf a0 = load_frag_bf16(A0 + kb, half);
    v16bf a1 = load_frag_bf16(A1 + kb, half);
#pragma unroll
    for (int t = 0; t < 4; ++t) {
      v16bf bfrag = load_frag_bf16(Wr[t] + kb, half);
      acc0[t] = WMMA_BF16(a0, bfrag, acc0[t]);
      acc1[t] = WMMA_BF16(a1, bfrag, acc1[t]);
    }
  }

#pragma unroll
  for (int t = 0; t < 4; ++t) {
    const int col = n0 + t * 16 + nidx;
    const float bv = bias[col];
#pragma unroll
    for (int vr = 0; vr < 8; ++vr) {
      const int r0 = m0 + vr + 8 * half;        // C/D layout: M = vgpr + 8*half
      float x0 = acc0[t][vr] + bv;
      float x1 = acc1[t][vr] + bv;
      if (Cf) {
        Cf[(size_t)r0 * N + col]        = x0;
        Cf[(size_t)(r0 + 16) * N + col] = x1;
      } else {
        Cb[(size_t)r0 * N + col]        = f2bfu(x0);
        Cb[(size_t)(r0 + 16) * N + col] = f2bfu(x1);
      }
    }
  }
}

// ---------------------------------------------------------------------------
// Vb [B,S,512] (bf16) -> Vt [B, kv, d, S] (bf16): AV B-operand key-contiguous
// ---------------------------------------------------------------------------
__global__ void __launch_bounds__(256)
vtranspose_kernel(const uint16_t* __restrict__ Vb, uint16_t* __restrict__ Vt) {
  int o = blockIdx.x * blockDim.x + threadIdx.x;
  if (o >= BATCH * N_KV * D_K * SEQ) return;
  int s  = o & (SEQ - 1);
  int d  = (o >> 11) & (D_K - 1);
  int kv = (o >> 17) & (N_KV - 1);
  int b  = o >> 20;
  Vt[o] = Vb[(size_t)(b * SEQ + s) * KV_DIM + kv * D_K + d];
}

// ---------------------------------------------------------------------------
// Flash-style causal attention, one wave per (head, 16-row q block).
//   Qb [B,S,2048] bf16, Kb [B,S,512] bf16, Vt [B,kv,d,S] bf16
//   Oh [B,S,2048] bf16  (channel = (kv*4+g)*64 + d, matches reference order)
// ---------------------------------------------------------------------------
__global__ void __launch_bounds__(32)
attention_kernel(const uint16_t* __restrict__ Qb, const uint16_t* __restrict__ Kb,
                 const uint16_t* __restrict__ Vt, uint16_t* __restrict__ Oh) {
  __shared__ float Sbuf[16 * 32];
  __shared__ __align__(16) uint32_t Pbuf32[16 * 16];   // 16 rows x 32 bf16
  __shared__ float Arow[16];
  __shared__ float Mrow[16];
  __shared__ float Lrow[16];
  __shared__ float Linv[16];

  const int lane = threadIdx.x & 31;
  const int half = (lane >> 4) & 1;
  const int nidx = lane & 15;
  const int i  = blockIdx.x;                 // q block (0..127)
  const int h  = blockIdx.y;                 // 0..63
  const int b  = h >> 5;
  const int hh = h & 31;                     // kv*4 + g
  const int kv = hh >> 2;

  if (lane < 16) { Mrow[lane] = -3.0e38f; Lrow[lane] = 0.0f; }
  __syncthreads();

  // Q fragments: M = q rows, K = d (64 -> two 32-wide steps)
  const bf16_t* qrow =
      (const bf16_t*)Qb + (size_t)(b * SEQ + i * 16 + nidx) * D_MODEL + hh * D_K;
  v16bf qf0 = load_frag_bf16(qrow, half);
  v16bf qf1 = load_frag_bf16(qrow + 32, half);

  v8f Ot[4] = {};

  const int jmax = i >> 1;                   // causal: keys up to i*16+15
  for (int j = 0; j <= jmax; ++j) {
    // ---- scores S[16 q x 32 keys] = Q K^T / 8 -----------------------------
    v8f Ct[2] = {};
#pragma unroll
    for (int t = 0; t < 2; ++t) {
      const bf16_t* krow = (const bf16_t*)Kb +
          (size_t)(b * SEQ + j * 32 + t * 16 + nidx) * KV_DIM + kv * D_K;
      v16bf k0 = load_frag_bf16(krow, half);
      v16bf k1 = load_frag_bf16(krow + 32, half);
      Ct[t] = WMMA_BF16(qf0, k0, Ct[t]);
      Ct[t] = WMMA_BF16(qf1, k1, Ct[t]);
    }
    // ---- scale + causal mask, stash to LDS --------------------------------
#pragma unroll
    for (int t = 0; t < 2; ++t) {
      const int kcol = j * 32 + t * 16 + nidx;
#pragma unroll
      for (int vr = 0; vr < 8; ++vr) {
        const int rloc = vr + 8 * half;
        float s = Ct[t][vr] * 0.125f;        // 1/sqrt(64)
        if (kcol > i * 16 + rloc) s = -1.0e30f;
        Sbuf[rloc * 32 + t * 16 + nidx] = s;
      }
    }
    __syncthreads();
    // ---- online softmax: lanes 0-15 each own one q row --------------------
    if (lane < 16) {
      const int r = lane;
      float mold = Mrow[r];
      float mnew = mold;
#pragma unroll
      for (int c = 0; c < 32; ++c) mnew = fmaxf(mnew, Sbuf[r * 32 + c]);
      float alpha = __expf(mold - mnew);
      float sum = 0.0f;
#pragma unroll
      for (int c = 0; c < 16; ++c) {
        float p0 = __expf(Sbuf[r * 32 + 2 * c]     - mnew);
        float p1 = __expf(Sbuf[r * 32 + 2 * c + 1] - mnew);
        Pbuf32[r * 16 + c] = pack2bf(p0, p1);
        sum += p0 + p1;
      }
      Mrow[r] = mnew;
      Lrow[r] = Lrow[r] * alpha + sum;
      Arow[r] = alpha;
    }
    __syncthreads();
    // ---- O = O*alpha + P V ------------------------------------------------
    float al[8];
#pragma unroll
    for (int vr = 0; vr < 8; ++vr) al[vr] = Arow[vr + 8 * half];
    v16bf pf = load_frag_bf16((const bf16_t*)&Pbuf32[nidx * 16], half);
#pragma unroll
    for (int t = 0; t < 4; ++t) {
#pragma unroll
      for (int vr = 0; vr < 8; ++vr) Ot[t][vr] *= al[vr];
      const bf16_t* vrow = (const bf16_t*)Vt +
          ((size_t)(b * N_KV + kv) * D_K + t * 16 + nidx) * SEQ + j * 32;
      v16bf vf = load_frag_bf16(vrow, half);
      Ot[t] = WMMA_BF16(pf, vf, Ot[t]);
    }
    __syncthreads();                         // Sbuf/Pbuf/Arow reuse next j
  }

  if (lane < 16) Linv[lane] = 1.0f / Lrow[lane];
  __syncthreads();

#pragma unroll
  for (int t = 0; t < 4; ++t) {
#pragma unroll
    for (int vr = 0; vr < 8; ++vr) {
      const int rloc = vr + 8 * half;
      Oh[(size_t)(b * SEQ + i * 16 + rloc) * D_MODEL + hh * D_K + t * 16 + nidx] =
          f2bfu(Ot[t][vr] * Linv[rloc]);
    }
  }
}

// ---------------------------------------------------------------------------
extern "C" void kernel_launch(void* const* d_in, const int* in_sizes, int n_in,
                              void* d_out, int out_size, void* d_ws, size_t ws_size,
                              hipStream_t stream) {
  const float* q  = (const float*)d_in[0];
  const float* k  = (const float*)d_in[1];
  const float* v  = (const float*)d_in[2];
  /* d_in[3] = causal mask (bool) — regenerated analytically in-kernel */
  const float* Wq = (const float*)d_in[4];
  const float* bq = (const float*)d_in[5];
  const float* Wk = (const float*)d_in[6];
  const float* bk = (const float*)d_in[7];
  const float* Wv = (const float*)d_in[8];
  const float* bv = (const float*)d_in[9];
  const float* Wo = (const float*)d_in[10];
  const float* bo = (const float*)d_in[11];
  float* out = (float*)d_out;

  // Workspace layout (bytes). Sequential launches on one stream make the
  // two aliases safe: Vt overwrites vbf only after the V GEMM, Oh overwrites
  // qbf only after the Q GEMM.
  const size_t MB = 1024 * 1024;
  uint8_t* ws = (uint8_t*)d_ws;
  uint16_t* qbf = (uint16_t*)(ws);            // 16 MB  [4096,2048] bf16
  uint16_t* kbf = (uint16_t*)(ws + 16 * MB);  // 16 MB
  uint16_t* vbf = (uint16_t*)(ws + 32 * MB);  // 16 MB
  uint16_t* Wqb = (uint16_t*)(ws + 48 * MB);  //  8 MB  [2048,2048]
  uint16_t* Wkb = (uint16_t*)(ws + 56 * MB);  //  2 MB  [512,2048]
  uint16_t* Wvb = (uint16_t*)(ws + 58 * MB);  //  2 MB
  uint16_t* Wob = (uint16_t*)(ws + 60 * MB);  //  8 MB
  uint16_t* Qb  = (uint16_t*)(ws + 68 * MB);  // 16 MB  [4096,2048]
  uint16_t* Kb  = (uint16_t*)(ws + 84 * MB);  //  4 MB  [4096,512]
  uint16_t* Vb  = (uint16_t*)(ws + 88 * MB);  //  4 MB  [4096,512]
  uint16_t* Vt  = vbf;                        //  4 MB  [2,8,64,2048] (alias)
  uint16_t* Ohb = qbf;                        // 16 MB  [4096,2048]   (alias)

  auto cvt = [&](const float* src, uint16_t* dst, int n) {
    int n4 = n / 4;
    cvt_f32_bf16_kernel<<<(n4 + 255) / 256, 256, 0, stream>>>(src, (uint32_t*)dst, n4);
  };
  auto gemm = [&](const uint16_t* A, const uint16_t* W, const float* bias,
                  float* Cf, uint16_t* Cb, int M, int N, int K) {
    int tiles  = (M / 32) * (N / 64);
    int blocks = (tiles + 7) / 8;             // 8 waves / block
    gemm_bf16_kernel<<<blocks, 256, 0, stream>>>(
        (const bf16_t*)A, (const bf16_t*)W, bias, Cf, Cb, M, N, K);
  };

  // 1) bf16 conversions
  cvt(q,  qbf, MTOK * D_MODEL);
  cvt(k,  kbf, MTOK * D_MODEL);
  cvt(v,  vbf, MTOK * D_MODEL);
  cvt(Wq, Wqb, D_MODEL * D_MODEL);
  cvt(Wk, Wkb, KV_DIM * D_MODEL);
  cvt(Wv, Wvb, KV_DIM * D_MODEL);
  cvt(Wo, Wob, D_MODEL * D_MODEL);

  // 2) projections (bf16 out)
  gemm(qbf, Wqb, bq, nullptr, Qb, MTOK, D_MODEL, D_MODEL);
  gemm(kbf, Wkb, bk, nullptr, Kb, MTOK, KV_DIM, D_MODEL);
  gemm(vbf, Wvb, bv, nullptr, Vb, MTOK, KV_DIM, D_MODEL);

  // 3) V transpose for AV operand
  {
    int total = BATCH * N_KV * D_K * SEQ;
    vtranspose_kernel<<<(total + 255) / 256, 256, 0, stream>>>(Vb, Vt);
  }

  // 4) attention
  attention_kernel<<<dim3(SEQ / 16, BATCH * N_HEADS), 32, 0, stream>>>(Qb, Kb, Vt, Ohb);

  // 5) output projection (fp32 out, with bias)
  gemm(Ohb, Wob, bo, out, nullptr, MTOK, D_MODEL, D_MODEL);
}
